// DynamicMaskHead_21260088115984
// MI455X (gfx1250) — matrix operand order repaired and checked by reading, not compiled, not run on hardware
//
#include <hip/hip_runtime.h>

// CondInst DynamicMaskHead on gfx1250 (MI455X), wave32, 3x WMMA chain per tile,
// 2 tiles stage-interleaved per iteration (fills all XDL hazard slots).
//
// out[n,p] = w2 . relu(w1 . relu(w0 . x_p + b0) + b1) + b2
//
// Rel-coord folding: relx = Ax - px*S. The affine part is folded into WMMA:
//   - B channels 8,9 carry raw (px, py) as f16 (exact: px<152, py<100),
//     precomputed per-pixel in d_ws next to the f16 feats (32 B/pixel).
//   - A0 K=8,9 entries are -S*w0[m][0], -S*w0[m][1].
//   - C0 bias is b0[m] + Ax*w0[m][0] + Ay*w0[m][1].
// => zero coordinate math in the hot loop.
//
// Layers 0/1: V_WMMA_F16_16X16X32_F16 (f16 C/D; lane N holds channels 0..7 in
// halves 0..7 = exactly the next B operand's low quad). Relu = 4x raw
// v_pk_max_num_f16 via inline asm. Layer 2: f32 accumulate, d2[0] NT-stored.
// A columns K>=valid are zero, so B slots for K>=valid (incl. lanes 16-31)
// may hold finite garbage; b vectors are loop-carried so zero quads persist.

typedef __attribute__((ext_vector_type(16))) _Float16 v16h;
typedef __attribute__((ext_vector_type(8)))  _Float16 v8h;
typedef __attribute__((ext_vector_type(2)))  _Float16 v2h;
typedef __attribute__((ext_vector_type(8)))  float    v8f;
typedef __attribute__((ext_vector_type(4)))  unsigned v4u;

#if __has_builtin(__builtin_amdgcn_wmma_f16_16x16x32_f16)
#define F16ACC 1
#else
#define F16ACC 0
#endif

namespace {
constexpr int N_IMG   = 2;
constexpr int C_IN    = 8;
constexpr int H       = 100;
constexpr int W       = 152;
constexpr int N_INST  = 500;
constexpr int HW      = H * W;            // 15200 = 475 * 32
constexpr int NPARAMS = 169;
constexpr int SPLIT   = 4;
constexpr int WAVES   = 8;
constexpr int NSLOTS  = SPLIT * WAVES;    // 32 wave-slots per instance
constexpr int PAIRPIX = 32;               // 2 tiles of 16 pixels per iteration
constexpr int STRIDEP = NSLOTS * PAIRPIX; // 1024-pixel loop stride
constexpr int PIXH    = 16;               // halves per pixel in ws (32 B)
constexpr size_t WS_NEEDED = (size_t)N_IMG * HW * PIXH * 2;
}

// ---- pre-pass: [img][c][p] f32 -> [img][p][f0..f7, px, py, pad] f16 --------
__global__ __launch_bounds__(256)
void preconvert_feats_f16(const float* __restrict__ mf, _Float16* __restrict__ ws16)
{
    const int idx = blockIdx.x * blockDim.x + threadIdx.x;
    if (idx >= N_IMG * HW) return;
    const int img = idx / HW;
    const int p   = idx - img * HW;
    const int py  = p / W;
    const int px  = p - py * W;
    const float* __restrict__ src = mf + (size_t)img * C_IN * HW + p;
    v8h lo;
#pragma unroll
    for (int c = 0; c < C_IN; ++c) lo[c] = (_Float16)src[(size_t)c * HW];
    v8h hi = {};
    hi[0] = (_Float16)px;     // exact in f16 (< 2048)
    hi[1] = (_Float16)py;
    _Float16* dst = ws16 + (size_t)idx * PIXH;
    *reinterpret_cast<v8h*>(dst)     = lo;
    *reinterpret_cast<v8h*>(dst + 8) = hi;
}

__device__ __forceinline__ void store_low8(v16h& b, const v8h t)
{
#pragma unroll
    for (int j = 0; j < 8; ++j) b[j] = t[j];
}

// raw v_pk_max_num_f16 x, x, 0 (no canonicalize)
__device__ __forceinline__ v8h relu8h(const v8h t)
{
    v4u u = __builtin_bit_cast(v4u, t);
#pragma unroll
    for (int i = 0; i < 4; ++i) {
        unsigned r;
        asm("v_pk_max_num_f16 %0, %1, 0" : "=v"(r) : "v"(u[i]));
        u[i] = r;
    }
    return __builtin_bit_cast(v8h, u);
}

__device__ __forceinline__ v8h cvt8(const v8f d)
{
    v8h t;
#pragma unroll
    for (int j = 0; j < 8; ++j) t[j] = (_Float16)d[j];
    return t;
}

template <bool PRE16>
__global__ __launch_bounds__(256)
void condinst_mask_head_wmma(const float* __restrict__ mask_feats,
                             const _Float16* __restrict__ feats16,
                             const float* __restrict__ params,
                             const float* __restrict__ iloc,
                             const float* __restrict__ soi,
                             const int*   __restrict__ im_inds,
                             float*       __restrict__ out)
{
    const int n      = blockIdx.y;
    const int tid    = threadIdx.x;
    const int lane   = tid & 31;
    const int lane15 = lane & 15;
    const int wave   = __builtin_amdgcn_readfirstlane(tid >> 5);

    __shared__ float sp[NPARAMS];
    if (tid < NPARAMS) sp[tid] = params[(size_t)n * NPARAMS + tid];
    __syncthreads();

    const float inv_soi = 1.0f / soi[n];
    const float S  = 8.0f * inv_soi;
    const float Ax = (iloc[2 * n + 0] - 4.0f) * inv_soi;
    const float Ay = (iloc[2 * n + 1] - 4.0f) * inv_soi;
    const int   img = im_inds[n];
    const float*    __restrict__ fbf = mask_feats + (size_t)img * C_IN * HW;
    const _Float16* __restrict__ fbh = feats16 + (size_t)img * HW * PIXH;

    // ---- invariant A operands (channels: f0..f7, px, py) ----
    const int m = lane15;
    v16h a0 = {}, a1 = {}, a2 = {};
    if (m < 8) {
        if (lane < 16) {
#pragma unroll
            for (int k = 0; k < 8; ++k) a0[k] = (_Float16)sp[m * 10 + 2 + k];
#pragma unroll
            for (int k = 0; k < 8; ++k) a1[k] = (_Float16)sp[80 + m * 8 + k];
            // elements 8,9 (pair with B px,py): -S*w_relx, -S*w_rely
            a0[8] = (_Float16)(-S * sp[m * 10 + 0]);
            a0[9] = (_Float16)(-S * sp[m * 10 + 1]);
        }
    }
    if (lane == 0) {
#pragma unroll
        for (int k = 0; k < 8; ++k) a2[k] = (_Float16)sp[144 + k];
    }
#if F16ACC
    v8h ch0 = {}, ch1 = {};
    if (lane < 16) {
#pragma unroll
        for (int j = 0; j < 8; ++j) {
            ch0[j] = (_Float16)(sp[152 + j] + Ax * sp[j * 10 + 0] + Ay * sp[j * 10 + 1]);
            ch1[j] = (_Float16)sp[160 + j];
        }
    }
#else
    v8f c0 = {}, c1 = {};
    if (lane < 16) {
#pragma unroll
        for (int j = 0; j < 8; ++j) {
            c0[j] = sp[152 + j] + Ax * sp[j * 10 + 0] + Ay * sp[j * 10 + 1];
            c1[j] = sp[160 + j];
        }
    }
#endif
    v8f c2 = {};
    if (lane < 16) c2[0] = sp[168];

    // ---- loop-carried B operands: upper quads written ONCE (stay zero) ----
    v16h b0A = {}, b1A = {}, b2A = {};
    v16h b0B = {}, b1B = {}, b2B = {};

    // both tile chains advance together: every WMMA result is consumed only
    // after another WMMA + 4 VALU have issued -> no hazard nops.
    auto pair = [&](const v8h fA, const v2h pA, const v8h fB, const v2h pB,
                    float& rA, float& rB) {
        store_low8(b0A, fA); b0A[8] = pA[0]; b0A[9] = pA[1];
        store_low8(b0B, fB); b0B[8] = pB[0]; b0B[9] = pB[1];
#if F16ACC
        const v8h d0A = __builtin_amdgcn_wmma_f16_16x16x32_f16(
            false, a0, false, b0A, (short)0, ch0, false, false);
        const v8h d0B = __builtin_amdgcn_wmma_f16_16x16x32_f16(
            false, a0, false, b0B, (short)0, ch0, false, false);
        store_low8(b1A, relu8h(d0A));
        store_low8(b1B, relu8h(d0B));
        const v8h d1A = __builtin_amdgcn_wmma_f16_16x16x32_f16(
            false, a1, false, b1A, (short)0, ch1, false, false);
        const v8h d1B = __builtin_amdgcn_wmma_f16_16x16x32_f16(
            false, a1, false, b1B, (short)0, ch1, false, false);
        store_low8(b2A, relu8h(d1A));
        store_low8(b2B, relu8h(d1B));
#else
        const v8f d0A = __builtin_amdgcn_wmma_f32_16x16x32_f16(
            false, a0, false, b0A, (short)0, c0, false, false);
        const v8f d0B = __builtin_amdgcn_wmma_f32_16x16x32_f16(
            false, a0, false, b0B, (short)0, c0, false, false);
        store_low8(b1A, relu8h(cvt8(d0A)));
        store_low8(b1B, relu8h(cvt8(d0B)));
        const v8f d1A = __builtin_amdgcn_wmma_f32_16x16x32_f16(
            false, a1, false, b1A, (short)0, c1, false, false);
        const v8f d1B = __builtin_amdgcn_wmma_f32_16x16x32_f16(
            false, a1, false, b1B, (short)0, c1, false, false);
        store_low8(b2A, relu8h(cvt8(d1A)));
        store_low8(b2B, relu8h(cvt8(d1B)));
#endif
        const v8f d2A = __builtin_amdgcn_wmma_f32_16x16x32_f16(
            false, a2, false, b2A, (short)0, c2, false, false);
        const v8f d2B = __builtin_amdgcn_wmma_f32_16x16x32_f16(
            false, a2, false, b2B, (short)0, c2, false, false);
        rA = d2A[0];
        rB = d2B[0];
    };

    int p0 = (blockIdx.x * WAVES + wave) * PAIRPIX;   // uniform (SGPR)
    float* __restrict__ op = out + (size_t)n * HW + p0;

    if (PRE16) {
        // software pipeline: next iteration's loads issued before this chain
        const _Float16* fl = fbh + (size_t)p0 * PIXH + (size_t)lane15 * PIXH;
        v8h fA = *reinterpret_cast<const v8h*>(fl);
        v2h pA = *reinterpret_cast<const v2h*>(fl + 8);
        v8h fB = *reinterpret_cast<const v8h*>(fl + 16 * PIXH);
        v2h pB = *reinterpret_cast<const v2h*>(fl + 16 * PIXH + 8);
        while (true) {
            const int p0n = p0 + STRIDEP;
            const bool more = p0n < HW;                       // uniform
            const _Float16* fln = fbh + (size_t)(more ? p0n : p0) * PIXH
                                      + (size_t)lane15 * PIXH;
            const v8h fA2 = *reinterpret_cast<const v8h*>(fln);
            const v2h pA2 = *reinterpret_cast<const v2h*>(fln + 8);
            const v8h fB2 = *reinterpret_cast<const v8h*>(fln + 16 * PIXH);
            const v2h pB2 = *reinterpret_cast<const v2h*>(fln + 16 * PIXH + 8);

            float rA, rB;
            pair(fA, pA, fB, pB, rA, rB);
            if (lane < 16) {
                __builtin_nontemporal_store(rA, op + lane15);
                __builtin_nontemporal_store(rB, op + 16 + lane15);
            }
            if (!more) break;
            fA = fA2; pA = pA2; fB = fB2; pB = pB2;
            p0 = p0n; op += STRIDEP;
        }
    } else {
        for (; p0 < HW; p0 += STRIDEP, op += STRIDEP) {
            // cold fallback: compute (px,py) + load f32 feats directly
            const int py0 = (int)((unsigned)p0 / (unsigned)W);
            const int px0 = p0 - py0 * W;
            int px = px0 + lane15;
            const int ov = (px >= W) ? 1 : 0;
            px -= ov ? W : 0;
            int pxb = px + 16;
            const int ovb = (pxb >= W) ? 1 : 0;
            pxb -= ovb ? W : 0;
            v2h pA, pB;
            pA[0] = (_Float16)px;  pA[1] = (_Float16)(py0 + ov);
            pB[0] = (_Float16)pxb; pB[1] = (_Float16)(py0 + ov + ovb);
            v8h fA, fB;
#pragma unroll
            for (int k = 0; k < 8; ++k) {
                fA[k] = (_Float16)fbf[(size_t)k * HW + p0 + lane15];
                fB[k] = (_Float16)fbf[(size_t)k * HW + p0 + 16 + lane15];
            }
            float rA, rB;
            pair(fA, pA, fB, pB, rA, rB);
            if (lane < 16) {
                __builtin_nontemporal_store(rA, op + lane15);
                __builtin_nontemporal_store(rB, op + 16 + lane15);
            }
        }
    }
}

extern "C" void kernel_launch(void* const* d_in, const int* in_sizes, int n_in,
                              void* d_out, int out_size, void* d_ws, size_t ws_size,
                              hipStream_t stream) {
    (void)in_sizes; (void)n_in; (void)out_size;
    const float* mask_feats = (const float*)d_in[0];
    const float* params     = (const float*)d_in[1];
    const float* iloc       = (const float*)d_in[2];
    const float* soi        = (const float*)d_in[3];
    const int*   im_inds    = (const int*)d_in[4];
    float*       out        = (float*)d_out;

    dim3 grid(SPLIT, N_INST);
    dim3 block(WAVES * 32);

    if (d_ws != nullptr && ws_size >= WS_NEEDED) {
        _Float16* ws16 = (_Float16*)d_ws;
        preconvert_feats_f16<<<(N_IMG * HW + 255) / 256, 256, 0, stream>>>(
            mask_feats, ws16);
        condinst_mask_head_wmma<true><<<grid, block, 0, stream>>>(
            mask_feats, ws16, params, iloc, soi, im_inds, out);
    } else {
        condinst_mask_head_wmma<false><<<grid, block, 0, stream>>>(
            mask_feats, nullptr, params, iloc, soi, im_inds, out);
    }
}
